// Model_7876970021388
// MI455X (gfx1250) — matrix operand order repaired and checked by reading, not compiled, not run on hardware
//
#include <hip/hip_runtime.h>
#include <math.h>

typedef __attribute__((ext_vector_type(16))) _Float16 v16h;
typedef __attribute__((ext_vector_type(8)))  float    v8f;

#define H 128

__device__ __forceinline__ float lrelu_f(float x){ return x > 0.f ? x : 0.1f*x; }
__device__ __forceinline__ float sigm_f(float x){ return 1.f/(1.f + __expf(-x)); }

// monotonic float<->uint for atomicMax on signed floats
__device__ __forceinline__ unsigned ordf(float f){
  unsigned u = __float_as_uint(f);
  return (u & 0x80000000u) ? ~u : (u | 0x80000000u);
}
__device__ __forceinline__ float unordf(unsigned u){
  u = (u & 0x80000000u) ? (u & 0x7fffffffu) : ~u;
  return __uint_as_float(u);
}

// ---------------------------------------------------------------------------
// WMMA GEMM: C[M,N] = act( A[M,K](lda) @ W + bias (+C if accum) )
// W row-major [K,N] (wtrans=0) or [N,K] (wtrans=1, i.e. x @ W.T)
// REQUIREMENT: M % 16 == 0 (true for every GEMM in this model).
// block = 256 threads = 8 wave32; block tile = 16 x 128; grid (M/16, N/128)
// LDS tiles are K-contiguous per row so each WMMA fragment VGPR (an adjacent
// K-pair of f16) is a single aligned ds_load_b32.
// ---------------------------------------------------------------------------
__global__ void __launch_bounds__(256)
k_gemm(const float* __restrict__ A, int lda,
       const float* __restrict__ W, const float* __restrict__ bias,
       float* __restrict__ C,
       int M, int N, int K, int wtrans, int act, int accum)
{
  __shared__ _Float16 sA[16][34];    // [m][k]   row = 68B (4B aligned)
  __shared__ _Float16 sB[128][34];   // [n][k]   (W tile stored transposed)
  const int tid  = threadIdx.x;
  const int lane = tid & 31;
  const int wv   = tid >> 5;     // 0..7 -> N sub-tile
  const int half = lane >> 4;    // 0/1
  const int l16  = lane & 15;
  const int m0 = blockIdx.x * 16;
  const int n0 = blockIdx.y * 128;

  v8f acc = {};
  const int nch = (K + 31) >> 5;
  for (int c = 0; c < nch; ++c) {
    const int k0 = c << 5;
    // ---- stage A tile 16x32 (branchless: clamp addr, select value) ----
    {
      int r0 = tid >> 5, cc = tid & 31;
      int k  = k0 + cc;
      int kk = k < K ? k : K - 1;
      bool ok = k < K;
#pragma unroll
      for (int rr = 0; rr < 16; rr += 8) {
        float v = A[(size_t)(m0 + r0 + rr)*lda + kk];
        sA[r0 + rr][cc] = (_Float16)(ok ? v : 0.f);
      }
    }
    // ---- stage W tile 32x128 into sB[n][k] ----
    if (!wtrans) {
      for (int i = tid; i < 32*128; i += 256) {
        int r = i >> 7, cc = i & 127;           // r=k-local, cc=n-local
        int k = k0 + r;
        int kk = k < K ? k : K - 1;
        float v = W[(size_t)kk*N + (n0 + cc)];  // coalesced over n
        sB[cc][r] = (_Float16)(k < K ? v : 0.f);
      }
    } else {
      for (int i = tid; i < 32*128; i += 256) {
        int r = i & 31, cc = i >> 5;            // r=k-local, cc=n-local
        int k = k0 + r;
        int kk = k < K ? k : K - 1;
        float v = W[(size_t)(n0 + cc)*K + kk];  // coalesced over k
        sB[cc][r] = (_Float16)(k < K ? v : 0.f);
      }
    }
    // prefetch next K-chunk (global_prefetch_b8; pure hint)
    if (c + 1 < nch) {
      __builtin_prefetch(A + (size_t)m0*lda + k0 + 32, 0, 1);
      __builtin_prefetch(wtrans ? W + (size_t)n0*K + k0 + 32
                                : W + (size_t)(k0 + 32)*N + n0, 0, 1);
    }
    __syncthreads();
    // ---- build fragments: 8 x ds_load_b32 each ----
    union { v16h h; unsigned u[8]; } av, bv;
    const unsigned* rowA = (const unsigned*)&sA[l16][0];
    const unsigned* rowB = (const unsigned*)&sB[wv*16 + l16][0];
#pragma unroll
    for (int v = 0; v < 8; ++v) {
      // A 16x32 f16 layout (ISA 7.12.2): lane=M; V0..3 K=2v+8*half, V4..7 K=16+2(v-4)+8*half
      int ka = (v < 4 ? 2*v : 16 + 2*(v-4)) + 8*half;   // even
      av.u[v] = rowA[ka >> 1];
      // B 32x16 f16 layout: lane=N; same K order as A
      int kb = (v < 4 ? 2*v : 16 + 2*(v-4)) + 8*half;   // even
      bv.u[v] = rowB[kb >> 1];
    }
    acc = __builtin_amdgcn_wmma_f32_16x16x32_f16(false, av.h, false, bv.h,
                                                 (short)0, acc, false, false);
    __syncthreads();
  }
  // C/D layout: VGPR r, lanes 0-15 -> M=r, lanes 16-31 -> M=8+r; N=lane&15
#pragma unroll
  for (int r = 0; r < 8; ++r) {
    int m = m0 + r + 8*half;
    int n = n0 + wv*16 + l16;
    size_t ci = (size_t)m*N + n;
    float v = acc[r];
    if (accum) v += C[ci];
    if (bias)  v += bias[n];
    if (act == 1) v = lrelu_f(v);
    else if (act == 2) v = sigm_f(v);
    C[ci] = v;
  }
}

// ---------------------------------------------------------------------------
// elementwise / scatter kernels
// ---------------------------------------------------------------------------
__global__ void k_fill(float* p, float v, long long n){
  long long i = (long long)blockIdx.x*blockDim.x + threadIdx.x;
  if (i < n) p[i] = v;
}

// out[seg[row]*C+f] += (w?w[row]:1) * x[row*C+f]
__global__ void k_seg_sum_rows(const float* __restrict__ x, const float* __restrict__ w,
                               const int* __restrict__ seg, float* __restrict__ out,
                               long long n, int C){
  long long i = (long long)blockIdx.x*blockDim.x + threadIdx.x;
  long long total = n*C;
  if (i >= total) return;
  int row = (int)(i / C), f = (int)(i % C);
  float v = x[i];
  if (w) v *= w[row];
  atomicAdd(&out[(size_t)seg[row]*C + f], v);
}

// out[dst[e]*ldo+ocol+f] += (ew?ew[e]:1) * h[src[e]*ldh+f]
__global__ void k_edge_agg(const float* __restrict__ h, int ldh,
                           const int* __restrict__ src, const int* __restrict__ dst,
                           const float* __restrict__ ew,
                           float* __restrict__ out, int ldo, int ocol,
                           long long E, int C){
  long long i = (long long)blockIdx.x*blockDim.x + threadIdx.x;
  long long total = E*C;
  if (i >= total) return;
  int e = (int)(i / C), f = (int)(i % C);
  float v = h[(size_t)src[e]*ldh + f];
  if (ew) v *= ew[e];
  atomicAdd(&out[(size_t)dst[e]*ldo + ocol + f], v);
}

__global__ void k_count(const int* __restrict__ dst, float* __restrict__ deg, int E){
  int e = blockIdx.x*blockDim.x + threadIdx.x;
  if (e < E) atomicAdd(&deg[dst[e]], 1.f);
}
__global__ void k_recipw(const int* __restrict__ dst, const float* __restrict__ deg,
                         float* __restrict__ w, int E){
  int e = blockIdx.x*blockDim.x + threadIdx.x;
  if (e < E) w[e] = 1.f / deg[dst[e]];
}
__global__ void k_gather_rows(const float* __restrict__ x, const int* __restrict__ idx,
                              float* __restrict__ out, long long n, int C){
  long long i = (long long)blockIdx.x*blockDim.x + threadIdx.x;
  long long total = n*C;
  if (i >= total) return;
  int row = (int)(i / C), f = (int)(i % C);
  out[i] = x[(size_t)idx[row]*C + f];
}
__global__ void k_mix(const float* __restrict__ z, const float* __restrict__ h,
                      const float* __restrict__ t, float* __restrict__ out, long long n){
  long long i = (long long)blockIdx.x*blockDim.x + threadIdx.x;
  if (i >= n) return;
  float zz = z[i];
  out[i] = zz*h[i] + (1.f - zz)*t[i];
}
// GRUCell combine: gi/gh are [rows,384], h/hn are [rows,128]
__global__ void k_gru(const float* __restrict__ gi, const float* __restrict__ gh,
                      const float* __restrict__ h, float* __restrict__ hn, long long rows){
  long long i = (long long)blockIdx.x*blockDim.x + threadIdx.x;
  long long total = rows * H;
  if (i >= total) return;
  int r = (int)(i / H), f = (int)(i % H);
  const float* gir = gi + (size_t)r*3*H;
  const float* ghr = gh + (size_t)r*3*H;
  float rg = sigm_f(gir[f]       + ghr[f]);
  float zg = sigm_f(gir[H+f]     + ghr[H+f]);
  float ng = tanhf (gir[2*H+f] + rg*ghr[2*H+f]);
  hn[i] = (1.f - zg)*ng + zg*h[i];
}
// segment max via monotonic-uint atomicMax;  s_eff = a*s + b
__global__ void k_seg_max(const float* __restrict__ s, float a, float b,
                          const int* __restrict__ seg, unsigned* __restrict__ m, int n){
  int i = blockIdx.x*blockDim.x + threadIdx.x;
  if (i < n) atomicMax(&m[seg[i]], ordf(a*s[i] + b));
}
__global__ void k_seg_expsum(const float* __restrict__ s, float a, float b,
                             const int* __restrict__ seg, const unsigned* __restrict__ m,
                             float* __restrict__ e, float* __restrict__ d, int n){
  int i = blockIdx.x*blockDim.x + threadIdx.x;
  if (i >= n) return;
  int g = seg[i];
  float ev = __expf(a*s[i] + b - unordf(m[g]));
  e[i] = ev;
  atomicAdd(&d[g], ev);
}
__global__ void k_seg_norm(const float* __restrict__ e, const float* __restrict__ d,
                           const int* __restrict__ seg, float* __restrict__ w, int n){
  int i = blockIdx.x*blockDim.x + threadIdx.x;
  if (i < n) w[i] = e[i] / d[seg[i]];
}
// score[v] = dot(qs[batch[v]], kv[v]) / sqrt(H);  one wave32 per row
__global__ void k_dot(const float* __restrict__ qs, const float* __restrict__ kv,
                      const int* __restrict__ batch, float* __restrict__ score, int n){
  int w = (blockIdx.x*blockDim.x + threadIdx.x) >> 5;
  int lane = threadIdx.x & 31;
  if (w >= n) return;
  const float* kr = kv + (size_t)w*H;
  const float* qr = qs + (size_t)batch[w]*H;
  float s = 0.f;
  for (int f = lane; f < H; f += 32) s += kr[f]*qr[f];
  for (int off = 16; off; off >>= 1) s += __shfl_xor(s, off, 32);
  if (lane == 0) score[w] = s * 0.08838834764831845f;   // 1/sqrt(128)
}
// out[p] = sqrt(sum_f (A[ia[p],f] - Bm[ib[p],f] + 1e-6)^2); one wave per pair
__global__ void k_pairdist(const float* __restrict__ A, const float* __restrict__ Bm,
                           const int* __restrict__ ia, const int* __restrict__ ib,
                           float* __restrict__ out, int P){
  int p = (blockIdx.x*blockDim.x + threadIdx.x) >> 5;
  int lane = threadIdx.x & 31;
  if (p >= P) return;
  const float* a = A  + (size_t)ia[p]*H;
  const float* b = Bm + (size_t)ib[p]*H;
  float s = 0.f;
  for (int f = lane; f < H; f += 32) { float d = a[f] - b[f] + 1e-6f; s += d*d; }
  for (int off = 16; off; off >>= 1) s += __shfl_xor(s, off, 32);
  if (lane == 0) out[p] = sqrtf(s);
}

// ---------------------------------------------------------------------------
extern "C" void kernel_launch(void* const* d_in, const int* in_sizes, int n_in,
                              void* d_out, int out_size, void* d_ws, size_t ws_size,
                              hipStream_t stream)
{
  (void)n_in; (void)out_size; (void)ws_size;
  // ---- inputs (setup_inputs dict order) ----
  const float* vert_x   = (const float*)d_in[0];
  const float* site_x   = (const float*)d_in[1];
  const float* masf_x   = (const float*)d_in[2];
  const float* prot_att = (const float*)d_in[3];
  const float* prom_att = (const float*)d_in[4];
  const int* vert_src = (const int*)d_in[5];   const int* vert_dst = (const int*)d_in[6];
  const int* vert_bat = (const int*)d_in[7];
  const int* comp_src = (const int*)d_in[8];   const int* comp_dst = (const int*)d_in[9];
  const int* site_src = (const int*)d_in[10];  const int* site_dst = (const int*)d_in[11];
  const int* masf_src = (const int*)d_in[12];  const int* masf_dst = (const int*)d_in[13];
  const int* prot_src = (const int*)d_in[14];  const int* prot_dst = (const int*)d_in[15];
  const int* prom_src = (const int*)d_in[16];  const int* prom_dst = (const int*)d_in[17];
  const int* anch_src = (const int*)d_in[18];  const int* anch_dst = (const int*)d_in[19];
  const int* idx_ag_a = (const int*)d_in[20];  const int* idx_ag_g = (const int*)d_in[21];
  const int* idx_aa_0 = (const int*)d_in[22];  const int* idx_aa_1 = (const int*)d_in[23];
  const int* idx_gg_0 = (const int*)d_in[24];  const int* idx_gg_1 = (const int*)d_in[25];
  const float* W_emb = (const float*)d_in[26]; const float* b_emb = (const float*)d_in[27];
  const float* gW_main = (const float*)d_in[28]; const float* gW_s2m = (const float*)d_in[29];
  const float* gW_gm = (const float*)d_in[30];   const float* gb_gm = (const float*)d_in[31];
  const float* gW_q = (const float*)d_in[32];    const float* gW_k = (const float*)d_in[33];
  const float* gW_mg = (const float*)d_in[34];   const float* gW_gs = (const float*)d_in[35];
  const float* gb_gs = (const float*)d_in[36];
  const float* gm_Wih = (const float*)d_in[37];  const float* gm_Whh = (const float*)d_in[38];
  const float* gm_bih = (const float*)d_in[39];  const float* gm_bhh = (const float*)d_in[40];
  const float* gs_Wih = (const float*)d_in[41];  const float* gs_Whh = (const float*)d_in[42];
  const float* gs_bih = (const float*)d_in[43];  const float* gs_bhh = (const float*)d_in[44];
  const float* W_s0 = (const float*)d_in[45];  const float* b_s0 = (const float*)d_in[46];
  const float* W_s1 = (const float*)d_in[47];  const float* b_s1 = (const float*)d_in[48];
  const float* W_so = (const float*)d_in[49];  const float* b_so = (const float*)d_in[50];
  const float* W_m0 = (const float*)d_in[51];  const float* b_m0 = (const float*)d_in[52];
  const float* W_m1 = (const float*)d_in[53];  const float* b_m1 = (const float*)d_in[54];
  const float* W_mo = (const float*)d_in[55];  const float* b_mo = (const float*)d_in[56];
  const float* W_a0 = (const float*)d_in[57];  const float* b_a0 = (const float*)d_in[58];
  const float* W_a1 = (const float*)d_in[59];  const float* b_a1 = (const float*)d_in[60];
  const float* W_ao = (const float*)d_in[61];  const float* b_ao = (const float*)d_in[62];
  const float* W_ds = (const float*)d_in[63];  const float* b_ds = (const float*)d_in[64];
  const float* W_dc = (const float*)d_in[65];  const float* b_dc = (const float*)d_in[66];

  const int FV = 82, Fs = 256, Fm = 320, Bb = 256, NG = 4000, NA = 10000;
  const int Nv = in_sizes[0]/FV, Ns = in_sizes[1]/Fs, Nm = in_sizes[2]/Fm;
  const int Ep = in_sizes[3];
  const int Ev = in_sizes[5], Ec = in_sizes[8], Es = in_sizes[10], Em = in_sizes[12],
            Ea = in_sizes[18];
  const int Pag = in_sizes[20], Paa = in_sizes[22], Pgg = in_sizes[24];
  const int D  = in_sizes[28]/(H*H);

  float* outp = (float*)d_out;

  // ---- launch helpers ----
  auto blk1 = [](long long n){ return dim3((unsigned)((n + 255)/256)); };
  auto fill = [&](float* p, float v, long long n){ k_fill<<<blk1(n),256,0,stream>>>(p, v, n); };
  auto gemm = [&](const float* A, int lda, const float* W, int wtr, const float* bias,
                  float* Cc, int M, int N, int K, int act, int accum){
    dim3 g((M+15)/16, N/128);
    k_gemm<<<g,256,0,stream>>>(A, lda, W, bias, Cc, M, N, K, wtr, act, accum);
  };
  auto edge_agg = [&](const float* h, int ldh, const int* src, const int* dst,
                      const float* ew, float* out, int ldo, int ocol, int E, int C){
    k_edge_agg<<<blk1((long long)E*C),256,0,stream>>>(h, ldh, src, dst, ew, out, ldo, ocol,
                                                      (long long)E, C);
  };

  // ---- workspace bump allocator ----
  size_t off = 0;
  auto alloc = [&](size_t nf)->float*{
    float* p = (float*)d_ws + off; off += (nf + 255) & ~(size_t)255; return p;
  };
  // persistent
  float* vA  = alloc((size_t)Nv*H);   float* vB  = alloc((size_t)Nv*H);
  float* sA_ = alloc((size_t)Bb*H);   float* sB_ = alloc((size_t)Bb*H);
  float* group    = alloc((size_t)NG*H);
  float* site_out = alloc((size_t)Ns*H);
  float* masf_out = alloc((size_t)Nm*H);
  float* anch     = alloc((size_t)NA*2*H);
  float* anch_out = alloc((size_t)NA*H);
  float* pa       = alloc((size_t)NA*H);
  float* pg       = alloc((size_t)NG*H);
  size_t mark = off;

  float* vcur = vA, *vnext = vB, *scur = sA_, *snext = sB_;

  // ---- embedding + super init ----
  gemm(vert_x, FV, W_emb, 0, b_emb, vcur, Nv, H, FV, /*lrelu*/1, 0);
  fill(scur, 0.f, (long long)Bb*H);
  k_seg_sum_rows<<<blk1((long long)Nv*H),256,0,stream>>>(vcur, nullptr, vert_bat, scur,
                                                         (long long)Nv, H);

  // ---- GWM scratch ----
  float* agg   = alloc((size_t)Nv*H);
  float* hmain = alloc((size_t)Nv*H);
  float* ts    = alloc((size_t)Bb*H);
  float* trans = alloc((size_t)Nv*H);
  float* zb    = alloc((size_t)Nv*H);
  float* gi    = alloc((size_t)Nv*3*H);
  float* gh    = alloc((size_t)Nv*3*H);
  float* qs    = alloc((size_t)Bb*H);
  float* kv    = alloc((size_t)Nv*H);
  float* score = alloc(Nv);
  float* ebuf  = alloc(Nv);
  float* attnw = alloc(Nv);
  float* mbuf  = alloc(Bb);
  float* dbuf  = alloc(Bb);
  float* pooled= alloc((size_t)Bb*H);
  float* msup  = alloc((size_t)Bb*H);
  float* zsb   = alloc((size_t)Bb*H);
  float* gis   = alloc((size_t)Bb*3*H);
  float* ghs   = alloc((size_t)Bb*3*H);
  float* wec   = alloc(Ec);
  float* deg   = alloc(NG);

  for (int t = 0; t < D; ++t) {
    const float* Wm  = gW_main + (size_t)t*H*H;
    const float* Ws2 = gW_s2m  + (size_t)t*H*H;
    const float* Wgm = gW_gm   + (size_t)t*2*H*H;   // [2H,H] row-major
    const float* bgm = gb_gm   + (size_t)t*H;
    const float* Wq  = gW_q    + (size_t)t*H*H;
    const float* Wk  = gW_k    + (size_t)t*H*H;
    const float* Wmg = gW_mg   + (size_t)t*H*H;
    const float* Wgs = gW_gs   + (size_t)t*2*H*H;
    const float* bgs = gb_gs   + (size_t)t*H;

    // main update
    fill(agg, 0.f, (long long)Nv*H);
    edge_agg(vcur, H, vert_src, vert_dst, nullptr, agg, H, 0, Ev, H);
    gemm(agg,  H, Wm,  0, nullptr, hmain, Nv, H, H, 1, 0);          // LRELU, no bias
    gemm(scur, H, Ws2, 0, nullptr, ts,    Bb, H, H, 1, 0);
    k_gather_rows<<<blk1((long long)Nv*H),256,0,stream>>>(ts, vert_bat, trans, (long long)Nv, H);
    gemm(hmain, H, Wgm,           0, bgm,     zb, Nv, H, H, 0, 0);  // top half of gate W
    gemm(trans, H, Wgm + (size_t)H*H, 0, nullptr, zb, Nv, H, H, 2, 1); // +bottom, sigmoid
    k_mix<<<blk1((long long)Nv*H),256,0,stream>>>(zb, hmain, trans, zb, (long long)Nv*H);
    gemm(zb,   H, gm_Wih, 1, gm_bih, gi, Nv, 3*H, H, 0, 0);         // x @ Wih.T + bih
    gemm(vcur, H, gm_Whh, 1, gm_bhh, gh, Nv, 3*H, H, 0, 0);
    k_gru<<<blk1((long long)Nv*H),256,0,stream>>>(gi, gh, vcur, vnext, (long long)Nv);

    // main -> super attention (k_head = 1)
    gemm(scur,  H, Wq, 0, nullptr, qs, Bb, H, H, 0, 0);
    gemm(vnext, H, Wk, 0, nullptr, kv, Nv, H, H, 0, 0);
    k_dot<<<dim3((Nv+7)/8),256,0,stream>>>(qs, kv, vert_bat, score, Nv);
    fill(mbuf, 0.f, Bb);
    k_seg_max<<<blk1(Nv),256,0,stream>>>(score, 1.f, 0.f, vert_bat, (unsigned*)mbuf, Nv);
    fill(dbuf, 0.f, Bb);
    k_seg_expsum<<<blk1(Nv),256,0,stream>>>(score, 1.f, 0.f, vert_bat, (unsigned*)mbuf,
                                            ebuf, dbuf, Nv);
    k_seg_norm<<<blk1(Nv),256,0,stream>>>(ebuf, dbuf, vert_bat, attnw, Nv);
    fill(pooled, 0.f, (long long)Bb*H);
    k_seg_sum_rows<<<blk1((long long)Nv*H),256,0,stream>>>(vnext, attnw, vert_bat, pooled,
                                                           (long long)Nv, H);
    gemm(pooled, H, Wmg, 0, nullptr, msup, Bb, H, H, 1, 0);
    gemm(msup, H, Wgs,            0, bgs,     zsb, Bb, H, H, 0, 0);
    gemm(scur, H, Wgs + (size_t)H*H, 0, nullptr, zsb, Bb, H, H, 2, 1);
    k_mix<<<blk1((long long)Bb*H),256,0,stream>>>(zsb, msup, scur, zsb, (long long)Bb*H);
    gemm(zsb,  H, gs_Wih, 1, gs_bih, gis, Bb, 3*H, H, 0, 0);
    gemm(scur, H, gs_Whh, 1, gs_bhh, ghs, Bb, 3*H, H, 0, 0);
    k_gru<<<blk1((long long)Bb*H),256,0,stream>>>(gis, ghs, scur, snext, (long long)Bb);

    { float* tmp = vcur; vcur = vnext; vnext = tmp; }
    { float* tmp = scur; scur = snext; snext = tmp; }
  }

  // ---- Group module: softmax(zeros) == 1/deg ----
  fill(deg, 0.f, NG);
  k_count<<<blk1(Ec),256,0,stream>>>(comp_dst, deg, Ec);
  k_recipw<<<blk1(Ec),256,0,stream>>>(comp_dst, deg, wec, Ec);
  fill(group, 0.f, (long long)NG*H);
  edge_agg(vcur, H, comp_src, comp_dst, wec, group, H, 0, Ec, H);

  // ---- conv stacks (dense-concat skip folded into split-K accumulation) ----
  off = mark;   // release GWM scratch
  auto conv_stack = [&](const float* x, int Fx, const int* src, const int* dst, int E, int Nn,
                        const float* W0, const float* b0, const float* W1, const float* b1,
                        const float* Wo, const float* bo, float* out){
    size_t save = off;
    float* agg0 = alloc((size_t)Nn*Fx);
    float* h1   = alloc((size_t)Nn*H);
    float* agg1 = agg0;                       // reuse (Fx >= 2H in all stacks)
    float* h2   = agg0 + (size_t)Nn*H;
    fill(agg0, 0.f, (long long)Nn*Fx);
    edge_agg(x, Fx, src, dst, nullptr, agg0, Fx, 0, E, Fx);
    gemm(agg0, Fx, W0, 0, b0, h1, Nn, H, Fx, 1, 0);
    fill(agg1, 0.f, (long long)Nn*H);
    edge_agg(h1, H, src, dst, nullptr, agg1, H, 0, E, H);
    gemm(agg1, H, W1, 0, b1, h2, Nn, H, H, 1, 0);
    // out = LRELU(x@Wo[0:Fx] + h1@Wo[Fx:Fx+H] + h2@Wo[Fx+H:] + bo)
    gemm(x,  Fx, Wo,                     0, bo,      out, Nn, H, Fx, 0, 0);
    gemm(h1, H,  Wo + (size_t)Fx*H,      0, nullptr, out, Nn, H, H,  0, 1);
    gemm(h2, H,  Wo + (size_t)(Fx+H)*H,  0, nullptr, out, Nn, H, H,  1, 1);
    off = save;
  };
  conv_stack(site_x, Fs, site_src, site_dst, Es, Ns, W_s0, b_s0, W_s1, b_s1, W_so, b_so, site_out);
  conv_stack(masf_x, Fm, masf_src, masf_dst, Em, Nm, W_m0, b_m0, W_m1, b_m1, W_mo, b_mo, masf_out);

  // ---- Anchor module: scatter_softmax(6 - dist) weighted aggregation ----
  float* mN = alloc(NA);
  float* dN = alloc(NA);
  float* eE = alloc(Ep);
  float* wEdge = alloc(Ep);
  fill(anch, 0.f, (long long)NA*2*H);
  // prot -> cols [0,H)
  fill(mN, 0.f, NA);
  k_seg_max<<<blk1(Ep),256,0,stream>>>(prot_att, -1.f, 6.f, prot_dst, (unsigned*)mN, Ep);
  fill(dN, 0.f, NA);
  k_seg_expsum<<<blk1(Ep),256,0,stream>>>(prot_att, -1.f, 6.f, prot_dst, (unsigned*)mN,
                                          eE, dN, Ep);
  k_seg_norm<<<blk1(Ep),256,0,stream>>>(eE, dN, prot_dst, wEdge, Ep);
  edge_agg(site_out, H, prot_src, prot_dst, wEdge, anch, 2*H, 0, Ep, H);
  // prom -> cols [H,2H)
  fill(mN, 0.f, NA);
  k_seg_max<<<blk1(Ep),256,0,stream>>>(prom_att, -1.f, 6.f, prom_dst, (unsigned*)mN, Ep);
  fill(dN, 0.f, NA);
  k_seg_expsum<<<blk1(Ep),256,0,stream>>>(prom_att, -1.f, 6.f, prom_dst, (unsigned*)mN,
                                          eE, dN, Ep);
  k_seg_norm<<<blk1(Ep),256,0,stream>>>(eE, dN, prom_dst, wEdge, Ep);
  edge_agg(masf_out, H, prom_src, prom_dst, wEdge, anch, 2*H, H, Ep, H);

  // ---- AnchorAnchor conv stack ----
  conv_stack(anch, 2*H, anch_src, anch_dst, Ea, NA, W_a0, b_a0, W_a1, b_a1, W_ao, b_ao, anch_out);

  // ---- Distance prediction ----
  gemm(anch_out, H, W_ds, 0, b_ds, pa, NA, H, H, 1, 0);
  gemm(group,    H, W_dc, 0, b_dc, pg, NG, H, H, 1, 0);
  k_pairdist<<<dim3((Pag+7)/8),256,0,stream>>>(pa, pg, idx_ag_a, idx_ag_g, outp, Pag);
  k_pairdist<<<dim3((Paa+7)/8),256,0,stream>>>(pa, pa, idx_aa_0, idx_aa_1, outp + Pag, Paa);
  k_pairdist<<<dim3((Pgg+7)/8),256,0,stream>>>(pg, pg, idx_gg_0, idx_gg_1, outp + Pag + Paa, Pgg);
}
